// DiseaseGNN_8005819040544
// MI455X (gfx1250) — matrix operand order repaired and checked by reading, not compiled, not run on hardware
//
#include <hip/hip_runtime.h>
#include <hip/hip_bf16.h>

#define N_NODES 100000
#define N_EDGES 3200000
#define F_INC   256
#define H1C     64
#define H2C     32
#define NCLS    1000

typedef __bf16 bf16_t;
typedef __attribute__((ext_vector_type(16))) __bf16 v16bf;
typedef __attribute__((ext_vector_type(8)))  float  v8f;

// pack two f32 into a (lo,hi) bf16 pair with one v_perm_b32:
// result bytes [3:0] = [u1.b3, u1.b2, u0.b3, u0.b2]
static __device__ __forceinline__ unsigned pk2bf(float f1, float f0) {
  return __builtin_amdgcn_perm(__float_as_uint(f1), __float_as_uint(f0), 0x07060302u);
}

static __device__ __forceinline__ void atomAddF(float* p, float v) {
  unsafeAtomicAdd(p, v);   // global_atomic_add_f32 (no CAS loop)
}

// ---------------- small utility kernels ----------------

__global__ void k_zero_u32(unsigned* __restrict__ p, int n) {
  int i = blockIdx.x * blockDim.x + threadIdx.x;
  if (i < n) p[i] = 0u;
}

__global__ void k_degree(const int* __restrict__ dst, unsigned* __restrict__ cnt, int e) {
  int i = blockIdx.x * blockDim.x + threadIdx.x;
  if (i < e) atomicAdd(&cnt[dst[i]], 1u);
}

// in-place: buffer holds uint counts, rewrite as dinv = rsqrt(1 + deg)
__global__ void k_dinv(float* __restrict__ d, int n) {
  int i = blockIdx.x * blockDim.x + threadIdx.x;
  if (i < n) {
    unsigned c = reinterpret_cast<const unsigned*>(d)[i];
    d[i] = rsqrtf(1.0f + (float)c);
  }
}

__global__ void k_coef(const int* __restrict__ src, const int* __restrict__ dst,
                       const float* __restrict__ dinv, float* __restrict__ coef, int e) {
  int i = blockIdx.x * blockDim.x + threadIdx.x;
  if (i < e) coef[i] = dinv[src[i]] * dinv[dst[i]];
}

// ---------------- WMMA GEMM with fused self-loop epilogue ----------------
// xw = X @ W  (M x KDIM) @ (KDIM x NOUT);  agg = xw * dinv^2 (self-loop init)
// W is pre-swizzled into LDS in exact B-fragment order so each fragment is one
// aligned 32-byte LDS read (2 x ds_load_b128) per lane.
template <int KDIM, int NOUT>
__global__ __launch_bounds__(256)
void k_gemm_selfloop(const float* __restrict__ X, const float* __restrict__ W,
                     const float* __restrict__ dinv,
                     float* __restrict__ xw, float* __restrict__ agg, int M) {
  constexpr int NT = NOUT / 16;            // 16x16 output tiles along N
  constexpr int KT = KDIM / 32;            // K-steps of 32
  constexpr int NWORDS = KDIM * NOUT / 2;  // packed bf16 pairs
  __shared__ unsigned sW[NWORDS];

  const int tid = threadIdx.x;
  // Fill LDS: word ((kt*NT+nt)*32 + lane)*8 + w  holds bf16 pair
  // (K = kb+2w, kb+2w+1) of column (lane&15)+16*nt, kb = 32kt + 16*(lane>>4).
  for (int i = tid; i < NWORDS; i += 256) {
    const int slot   = i >> 8;        // kt*NT + nt  (256 words per fragment set)
    const int within = i & 255;
    const int l  = within >> 3;       // lane
    const int w  = within & 7;        // word within fragment
    const int kt = slot / NT;
    const int nt = slot % NT;
    const int col = (l & 15) + nt * 16;
    const int kb  = kt * 32 + (l >> 4) * 16 + 2 * w;
    sW[i] = pk2bf(W[(kb + 1) * NOUT + col], W[kb * NOUT + col]);
  }
  __syncthreads();

  const int lane  = tid & 31;
  const int wave  = tid >> 5;
  const int mtile = blockIdx.x * 8 + wave;   // one 16-row strip per wave
  if (mtile * 16 >= M) return;

  const int kh = lane >> 4;                  // lane-half selects K sub-block
  const float* __restrict__ xrow = X + (size_t)(mtile * 16 + (lane & 15)) * KDIM;

  v8f acc[NT] = {};

  union AV { unsigned u[8]; v16bf v; };

#pragma unroll
  for (int kt = 0; kt < KT; ++kt) {
    // A fragment: lane holds K = {32kt+16kh+0..15}
    const int k0 = kt * 32 + kh * 16;
    const int k1 = k0 + 8;
    const float4 fa = *reinterpret_cast<const float4*>(xrow + k0);
    const float4 fb = *reinterpret_cast<const float4*>(xrow + k0 + 4);
    const float4 fc = *reinterpret_cast<const float4*>(xrow + k1);
    const float4 fd = *reinterpret_cast<const float4*>(xrow + k1 + 4);
    AV a;
    a.u[0] = pk2bf(fa.y, fa.x);  a.u[1] = pk2bf(fa.w, fa.z);
    a.u[2] = pk2bf(fb.y, fb.x);  a.u[3] = pk2bf(fb.w, fb.z);
    a.u[4] = pk2bf(fc.y, fc.x);  a.u[5] = pk2bf(fc.w, fc.z);
    a.u[6] = pk2bf(fd.y, fd.x);  a.u[7] = pk2bf(fd.w, fd.z);

#pragma unroll
    for (int nt = 0; nt < NT; ++nt) {
      const v16bf b = *reinterpret_cast<const v16bf*>(
          &sW[(((kt * NT) + nt) * 32 + lane) * 8]);   // 32B aligned
      acc[nt] = __builtin_amdgcn_wmma_f32_16x16x32_bf16(
          false, a.v, false, b, (short)0, acc[nt], false, false);
    }
  }

  // D layout: element r -> M = r + 8*kh, N = lane&15 (+16*nt)
#pragma unroll
  for (int r = 0; r < 8; ++r) {
    const int orow = mtile * 16 + r + 8 * kh;
    const float di = dinv[orow];
    const float dd = di * di;
#pragma unroll
    for (int nt = 0; nt < NT; ++nt) {
      const int ocol = (lane & 15) + nt * 16;
      const float v = acc[nt][r];
      xw [(size_t)orow * NOUT + ocol] = v;
      agg[(size_t)orow * NOUT + ocol] = v * dd;   // self-loop contribution
    }
  }
}

// ---------------- edge scatter: agg[dst] += xw[src] * coef ----------------
template <int H>
__global__ __launch_bounds__(256)
void k_scatter(const float* __restrict__ xw, const int* __restrict__ src,
               const int* __restrict__ dst, const float* __restrict__ coef,
               float* __restrict__ agg, int e) {
  constexpr int CH = H / 4;                      // float4 chunks per edge
  const long long idx = (long long)blockIdx.x * blockDim.x + threadIdx.x;
  if (idx >= (long long)e * CH) return;
  const int ed = (int)(idx / CH);
  const int c  = (int)(idx % CH) * 4;
  const float cf = coef[ed];
  const int s = src[ed], d = dst[ed];
  const float4 v = *reinterpret_cast<const float4*>(xw + (size_t)s * H + c);
  float* ap = agg + (size_t)d * H + c;
  atomAddF(ap + 0, v.x * cf);
  atomAddF(ap + 1, v.y * cf);
  atomAddF(ap + 2, v.z * cf);
  atomAddF(ap + 3, v.w * cf);
}

// in-place h = relu(agg + bias)   (H must be power of two; mask = H-1)
__global__ void k_bias_relu(float* __restrict__ a, const float* __restrict__ b,
                            long long total, int mask) {
  const long long i = (long long)blockIdx.x * blockDim.x + threadIdx.x;
  if (i < total) {
    const float v = a[i] + b[(int)(i & mask)];
    a[i] = v > 0.0f ? v : 0.0f;
  }
}

// fused layer-2 bias + relu + column-sum (mean pooling numerator)
__global__ void k_mean_relu(const float* __restrict__ agg2, const float* __restrict__ b2,
                            float* __restrict__ gsum, int n) {
  const int col = threadIdx.x & 31;
  const int rpb = blockDim.x >> 5;
  const float bias = b2[col];
  float s = 0.0f;
  for (int row = blockIdx.x * rpb + (threadIdx.x >> 5); row < n; row += gridDim.x * rpb) {
    const float v = agg2[(size_t)row * H2C + col] + bias;
    s += v > 0.0f ? v : 0.0f;
  }
  atomAddF(&gsum[col], s);
}

__global__ void k_fc(const float* __restrict__ gsum, const float* __restrict__ Wfc,
                     const float* __restrict__ bfc, float* __restrict__ out, float invN) {
  const int c = blockIdx.x * blockDim.x + threadIdx.x;
  if (c >= NCLS) return;
  float s = bfc[c];
#pragma unroll
  for (int k = 0; k < H2C; ++k) s += gsum[k] * invN * Wfc[k * NCLS + c];
  out[c] = s;
}

// ---------------- launch ----------------
extern "C" void kernel_launch(void* const* d_in, const int* in_sizes, int n_in,
                              void* d_out, int out_size, void* d_ws, size_t ws_size,
                              hipStream_t stream) {
  const float* x    = (const float*)d_in[0];
  const int*   esrc = (const int*)d_in[1];
  const int*   edst = (const int*)d_in[2];
  const float* W1   = (const float*)d_in[3];
  const float* b1   = (const float*)d_in[4];
  const float* W2   = (const float*)d_in[5];
  const float* b2   = (const float*)d_in[6];
  const float* Wfc  = (const float*)d_in[7];
  const float* bfc  = (const float*)d_in[8];
  float* out = (float*)d_out;

  float* ws   = (float*)d_ws;
  float* dinv = ws;                               // N floats (uint counts first)
  float* coef = dinv + N_NODES;                   // E floats
  float* xw1  = coef + N_EDGES;                   // N*64
  float* agg1 = xw1  + (size_t)N_NODES * H1C;     // N*64 (-> h1 after relu)
  float* xw2  = agg1 + (size_t)N_NODES * H1C;     // N*32
  float* agg2 = xw2  + (size_t)N_NODES * H2C;     // N*32
  float* gsum = agg2 + (size_t)N_NODES * H2C;     // 32

  // degrees + normalization
  k_zero_u32<<<(N_NODES + 255) / 256, 256, 0, stream>>>((unsigned*)dinv, N_NODES);
  k_zero_u32<<<1, 32, 0, stream>>>((unsigned*)gsum, 32);
  k_degree<<<(N_EDGES + 255) / 256, 256, 0, stream>>>(edst, (unsigned*)dinv, N_EDGES);
  k_dinv<<<(N_NODES + 255) / 256, 256, 0, stream>>>(dinv, N_NODES);
  k_coef<<<(N_EDGES + 255) / 256, 256, 0, stream>>>(esrc, edst, dinv, coef, N_EDGES);

  // layer 1: GEMM (WMMA bf16) + scatter + bias/relu
  k_gemm_selfloop<F_INC, H1C><<<(N_NODES + 127) / 128, 256, 0, stream>>>(
      x, W1, dinv, xw1, agg1, N_NODES);
  {
    const long long work = (long long)N_EDGES * (H1C / 4);
    k_scatter<H1C><<<(unsigned)((work + 255) / 256), 256, 0, stream>>>(
        xw1, esrc, edst, coef, agg1, N_EDGES);
    const long long tot = (long long)N_NODES * H1C;
    k_bias_relu<<<(unsigned)((tot + 255) / 256), 256, 0, stream>>>(agg1, b1, tot, H1C - 1);
  }

  // layer 2: GEMM (WMMA bf16) + scatter; relu fused into mean reduction
  k_gemm_selfloop<H1C, H2C><<<(N_NODES + 127) / 128, 256, 0, stream>>>(
      agg1, W2, dinv, xw2, agg2, N_NODES);
  {
    const long long work = (long long)N_EDGES * (H2C / 4);
    k_scatter<H2C><<<(unsigned)((work + 255) / 256), 256, 0, stream>>>(
        xw2, esrc, edst, coef, agg2, N_EDGES);
  }
  k_mean_relu<<<1024, 256, 0, stream>>>(agg2, b2, gsum, N_NODES);
  k_fc<<<(NCLS + 255) / 256, 256, 0, stream>>>(gsum, Wfc, bfc, out, 1.0f / (float)N_NODES);
}
